// StateEncoder_35777077576511
// MI455X (gfx1250) — matrix operand order repaired and checked by reading, not compiled, not run on hardware
//
#include <hip/hip_runtime.h>

typedef _Float16 half_t;
typedef __attribute__((ext_vector_type(16))) _Float16 v16h;
typedef __attribute__((ext_vector_type(8)))  _Float16 v8h;
typedef __attribute__((ext_vector_type(8)))  float    v8f;

#define N_MAX    64
#define D_MODEL  256
#define KF       8
#define OUT_ROWS 67          // 3 rod rows + 64 disk rows
// ln(10000)/8
#define OMEGA_LN 1.1512925465f

__device__ __forceinline__ void fourier_pair(int idx, int k, float& s, float& c) {
    float om  = __expf(-(float)k * OMEGA_LN);   // 10000^(-k/8)
    float ang = (float)idx * om;
    s = __sinf(ang);
    c = __cosf(ang);
}

__device__ __forceinline__ v16h gather_frag(const half_t* p0, const half_t* p1) {
    v8h lo = *(const v8h*)p0;
    v8h hi = *(const v8h*)p1;
    v16h r;
#pragma unroll
    for (int i = 0; i < 8; ++i) { r[i] = lo[i]; r[8 + i] = hi[i]; }
    return r;
}

// ---------------------------------------------------------------------------
// Disk kernel: rows = B*64, K = 29 (+bias row) padded to 32, one WMMA k-step.
// Block = 256 threads = 8 waves; each wave -> 16 rows x 256 cols.
// Block covers 128 rows = exactly 2 batch elements.
// Bias is folded into W^T row 29 with feature[29] = 1.0.
// ---------------------------------------------------------------------------
__global__ __launch_bounds__(256) void disk_kernel(
    const int*   __restrict__ state,
    const float* __restrict__ rod_id_embed,
    const float* __restrict__ disk_W,
    const float* __restrict__ disk_b,
    const float* __restrict__ padding_embedding,
    float*       __restrict__ out)
{
    __shared__ half_t sWt[D_MODEL][40];   // transposed W (+bias row), K padded to 32
    __shared__ half_t sA[8][16][40];      // per-wave A tiles (16 rows x 32 K)
    __shared__ float  sPad[D_MODEL];
    __shared__ int    sState[2][N_MAX];
    __shared__ float  sRodId[3][8];

    const int tid  = threadIdx.x;
    const int lane = tid & 31;
    const int w    = tid >> 5;

    // Cooperative: W^T (f32 -> f16) with bias folded into K-row 29
    for (int idx = tid; idx < 32 * D_MODEL; idx += 256) {
        int f = idx >> 8;           // 0..31 (K)
        int d = idx & 255;          // 0..255
        float v = (f < 29) ? disk_W[f * D_MODEL + d]
                           : ((f == 29) ? disk_b[d] : 0.f);
        sWt[d][f] = (half_t)v;
    }
    if (tid < D_MODEL) sPad[tid] = padding_embedding[tid];
    if (tid < 24) sRodId[tid >> 3][tid & 7] = rod_id_embed[tid];

    const long long row0b = (long long)blockIdx.x * 128;
    const int b0 = (int)(row0b >> 6);
    for (int idx = tid; idx < 2 * N_MAX; idx += 256)
        sState[idx >> 6][idx & 63] = state[(long long)b0 * N_MAX + idx];
    __syncthreads();

    const int wb    = (w * 16) >> 6;            // which of the 2 batch elems
    const int nbase = (w * 16) & 63;            // disk index base
    const int bidx  = b0 + wb;

    // ---- features: lanes 0..15, one disk row each ----
    if (lane < 16) {
        const int  n     = nbase + lane;
        const int  s     = sState[wb][n];
        const bool valid = (s != -1);
        const int  ss    = (s < 0) ? 0 : (s > 2 ? 2 : s);
        int above = 0, height = 0;
#pragma unroll 4
        for (int j = 0; j < N_MAX; ++j) {
            int sj = sState[wb][j];
            if (sj == ss && sj != -1) { height++; if (j < n) above++; }
        }
        float vf    = valid ? 1.f : 0.f;
        float fa    = (float)above;
        float fh    = (float)height;
        float depth = (fh > 1.f) ? (fa / fmaxf(fh - 1.f, 1.f)) : 0.f;
        depth *= vf;
        float is_top = (above == 0) ? vf : 0.f;
        float is_bot = (fa == fh - 1.f) ? vf : 0.f;

        half_t* f = sA[w][lane];
        f[0] = (half_t)1.f; f[1] = (half_t)0.f;
#pragma unroll
        for (int k = 0; k < KF; ++k) {
            float sn, cs; fourier_pair(n, k, sn, cs);
            f[2 + 2 * k] = (half_t)sn;
            f[3 + 2 * k] = (half_t)cs;
        }
        f[18] = (half_t)depth;
        f[19] = (half_t)is_top;
        f[20] = (half_t)is_bot;
#pragma unroll
        for (int i = 0; i < 8; ++i) f[21 + i] = (half_t)sRodId[ss][i];
        f[29] = (half_t)1.f;        // bias row selector
        f[30] = (half_t)0.f;
        f[31] = (half_t)0.f;
    }
    // same-wave LDS producer/consumer: DS ops are in-order per wave

    // ---- A fragment (ISA 16-bit 16x32 A layout) ----
    const int m   = lane & 15;
    const int kb  = (lane & 16) ? 8 : 0;
    const v16h a  = gather_frag(&sA[w][m][kb], &sA[w][m][kb + 16]);

    const int koff = (lane & 16) ? 8 : 0;
    const int mhi  = (lane & 16) ? 8 : 0;

    // per-row validity flags, hoisted out of the store loop
    bool vrow[8];
#pragma unroll
    for (int r = 0; r < 8; ++r)
        vrow[r] = (sState[wb][nbase + mhi + r] != -1);

    float* __restrict__ pout = out
        + (long long)bidx * (OUT_ROWS * D_MODEL)
        + (long long)(3 + nbase + mhi) * D_MODEL
        + (lane & 15);

    for (int nt = 0; nt < 16; ++nt) {
        const int ncol = nt * 16 + (lane & 15);
        const v16h bf = gather_frag(&sWt[ncol][koff], &sWt[ncol][koff + 16]);
        v8f c = {};
        c = __builtin_amdgcn_wmma_f32_16x16x32_f16(false, a, false, bf,
                                                   (short)0, c, false, false);
        const float pad = sPad[ncol];   // one LDS read per tile
#pragma unroll
        for (int r = 0; r < 8; ++r) {
            const float v = vrow[r] ? c[r] : pad;   // v_cndmask, no branch
            pout[r * D_MODEL + nt * 16] = v;
        }
    }
}

// ---------------------------------------------------------------------------
// Rod kernel: rows = B*3, K = 44 (+bias row) padded to 64, two WMMA k-steps.
// Block = 256 threads = 8 waves; each wave -> 16 rod rows x 256 cols.
// Bias is folded into W^T row 44 with feature[44] = 1.0.
// ---------------------------------------------------------------------------
__global__ __launch_bounds__(256) void rod_kernel(
    const int*   __restrict__ state,
    const float* __restrict__ rod_id_embed,
    const float* __restrict__ rod_W,
    const float* __restrict__ rod_b,
    const float* __restrict__ null_fourier,
    float*       __restrict__ out,
    long long rowsTotal)
{
    __shared__ half_t sWt[D_MODEL][72];   // transposed W (+bias row), K padded to 64
    __shared__ half_t sA[8][16][72];
    __shared__ float  sNull[16];
    __shared__ float  sRodId[3][8];

    const int tid  = threadIdx.x;
    const int lane = tid & 31;
    const int w    = tid >> 5;

    for (int idx = tid; idx < 64 * D_MODEL; idx += 256) {
        int f = idx >> 8;           // 0..63 (K)
        int d = idx & 255;
        float v = (f < 44) ? rod_W[f * D_MODEL + d]
                           : ((f == 44) ? rod_b[d] : 0.f);
        sWt[d][f] = (half_t)v;
    }
    if (tid < 16) sNull[tid] = null_fourier[tid];
    if (tid < 24) sRodId[tid >> 3][tid & 7] = rod_id_embed[tid];
    __syncthreads();

    const long long row0 = (long long)blockIdx.x * 128 + (long long)w * 16;

    // ---- features: lanes 0..15, one rod row each ----
    if (lane < 16) {
        const long long row = row0 + lane;
        half_t* f = sA[w][lane];
        if (row < rowsTotal) {
            const int b  = (int)(row / 3);
            const int rr = (int)(row - 3LL * b);
            const int* st = state + (long long)b * N_MAX;
            int count = 0, nvalid = 0, top = -1, bot = -1;
#pragma unroll 4
            for (int j = 0; j < N_MAX; ++j) {
                int sj = st[j];
                if (sj != -1) {
                    nvalid++;
                    if (sj == rr) { count++; if (top < 0) top = j; bot = j; }
                }
            }
            const bool empty = (count == 0);
            if (top < 0) top = 0;           // argmax of all-false = 0
            if (bot < 0) bot = N_MAX - 1;   // N-1 - argmax(rev all-false)
            const float len_norm = (float)count / (float)(nvalid > 1 ? nvalid : 1);

            f[0] = (half_t)0.f; f[1] = (half_t)1.f;
            f[2] = (half_t)len_norm;
            f[3] = (half_t)(empty ? 1.f : 0.f);
#pragma unroll
            for (int k = 0; k < KF; ++k) {
                float sn, cs; fourier_pair(top, k, sn, cs);
                f[4 + 2 * k] = (half_t)(empty ? sNull[2 * k]     : sn);
                f[5 + 2 * k] = (half_t)(empty ? sNull[2 * k + 1] : cs);
            }
#pragma unroll
            for (int k = 0; k < KF; ++k) {
                float sn, cs; fourier_pair(bot, k, sn, cs);
                f[20 + 2 * k] = (half_t)(empty ? sNull[2 * k]     : sn);
                f[21 + 2 * k] = (half_t)(empty ? sNull[2 * k + 1] : cs);
            }
#pragma unroll
            for (int i = 0; i < 8; ++i) f[36 + i] = (half_t)sRodId[rr][i];
            f[44] = (half_t)1.f;    // bias row selector
#pragma unroll
            for (int k = 45; k < 64; ++k) f[k] = (half_t)0.f;
        } else {
#pragma unroll
            for (int k = 0; k < 64; ++k) f[k] = (half_t)0.f;
        }
    }

    // ---- A fragments for both k-steps ----
    const int m   = lane & 15;
    const int kb  = (lane & 16) ? 8 : 0;
    const v16h a0 = gather_frag(&sA[w][m][kb],      &sA[w][m][kb + 16]);
    const v16h a1 = gather_frag(&sA[w][m][32 + kb], &sA[w][m][32 + kb + 16]);

    const int koff = (lane & 16) ? 8 : 0;
    const int mhi  = (lane & 16) ? 8 : 0;

    float* __restrict__ prow[8];
    bool ook[8];
#pragma unroll
    for (int r = 0; r < 8; ++r) {
        const long long row = row0 + mhi + r;
        const int b  = (int)(row / 3);
        const int rr = (int)(row - 3LL * b);
        prow[r] = out + (long long)b * (OUT_ROWS * D_MODEL)
                      + (long long)rr * D_MODEL + (lane & 15);
        ook[r]  = (row < rowsTotal);
    }

    for (int nt = 0; nt < 16; ++nt) {
        const int ncol = nt * 16 + (lane & 15);
        const v16h b0 = gather_frag(&sWt[ncol][koff],      &sWt[ncol][koff + 16]);
        const v16h b1 = gather_frag(&sWt[ncol][32 + koff], &sWt[ncol][32 + koff + 16]);
        v8f c = {};
        c = __builtin_amdgcn_wmma_f32_16x16x32_f16(false, a0, false, b0,
                                                   (short)0, c, false, false);
        c = __builtin_amdgcn_wmma_f32_16x16x32_f16(false, a1, false, b1,
                                                   (short)0, c, false, false);
#pragma unroll
        for (int r = 0; r < 8; ++r) {
            if (ook[r]) prow[r][nt * 16] = c[r];   // bias already in accumulator
        }
    }
}

// ---------------------------------------------------------------------------
extern "C" void kernel_launch(void* const* d_in, const int* in_sizes, int n_in,
                              void* d_out, int out_size, void* d_ws, size_t ws_size,
                              hipStream_t stream) {
    const int*   state  = (const int*)  d_in[0];
    const float* rod_id = (const float*)d_in[1];
    const float* disk_W = (const float*)d_in[2];
    const float* disk_b = (const float*)d_in[3];
    const float* rod_W  = (const float*)d_in[4];
    const float* rod_b  = (const float*)d_in[5];
    const float* null_f = (const float*)d_in[6];
    const float* pad_e  = (const float*)d_in[7];
    float* out = (float*)d_out;

    const int Btot = in_sizes[0] / N_MAX;               // 16384
    const long long diskRows = (long long)Btot * N_MAX; // 1,048,576
    const long long rodRows  = (long long)Btot * 3;     // 49,152

    const int diskBlocks = (int)(diskRows / 128);       // exact: 8192
    const int rodBlocks  = (int)((rodRows + 127) / 128);

    disk_kernel<<<dim3(diskBlocks), dim3(256), 0, stream>>>(
        state, rod_id, disk_W, disk_b, pad_e, out);
    rod_kernel<<<dim3(rodBlocks), dim3(256), 0, stream>>>(
        state, rod_id, rod_W, rod_b, null_f, out, rodRows);
}
